// CSAFM_9225589752073
// MI455X (gfx1250) — matrix operand order repaired and checked by reading, not compiled, not run on hardware
//
#include <hip/hip_runtime.h>
#include <hip/hip_bf16.h>
#include <math.h>

// ---------------- problem constants ----------------
#define BB    4
#define CC0   192
#define HW    16384      // 128*128
#define HEADS 4
#define CPG   48
#define C3    576        // 3*C
#define CMIX  432        // 9*CPG
#define KSPLIT 16        // K-split factor for q.k^T

typedef __bf16        bf16x16 __attribute__((ext_vector_type(16)));
typedef float         f32x8   __attribute__((ext_vector_type(8)));
typedef unsigned int  u32x8   __attribute__((ext_vector_type(8)));

__device__ inline unsigned int pack_bf16(float lo, float hi)
{
    unsigned short a = __builtin_bit_cast(unsigned short, (__bf16)lo);
    unsigned short b = __builtin_bit_cast(unsigned short, (__bf16)hi);
    return ((unsigned int)b << 16) | a;   // -> v_cvt_pk_bf16_f32
}

// =====================================================================
// Pointwise (1x1) conv as WMMA GEMM with LDS-shared B fragments:
//   OUT[b, m, p] = sum_c X[b, c, p] * W[m, c]   (+bias+relu, +addsrc)
// Block = 192 threads = 6 waves. The 6 waves cover 6 M-tiles (96 rows)
// over one 32-pixel N strip; the X chunk (32K x 32N) is staged once per
// K-chunk into LDS in bf16 WMMA-fragment order (double buffered).
// =====================================================================
template <int BIASRELU, int ADDSRC>
__global__ __launch_bounds__(192)
void pw_gemm_wmma(const float* __restrict__ X, const float* __restrict__ W,
                  const float* __restrict__ bias, const float* __restrict__ addsrc,
                  float* __restrict__ OUT, int Cin, int Cout)
{
    __shared__ unsigned int lds[2][512];   // [buf][t*256 + d*32 + lane]

    const int b    = blockIdx.x;
    const int nb   = blockIdx.z * 32;          // 32-pixel N strip
    const int wave = threadIdx.x >> 5;
    const int lane = threadIdx.x & 31;
    const int l15  = lane & 15;

    const float* Xb = X + (size_t)b * Cin * HW;
    float*       Ob = OUT + (size_t)b * Cout * HW;

    const int mbase = blockIdx.y * 96 + wave * 16;
    const int mrow  = mbase + l15;             // A row (weights)
    const int koffA = (lane < 16) ? 0 : 16;    // K split by lane half: contiguous k = 16h + s

    const int nchunks = Cin / 32;

    // ---- stage one K-chunk of X into LDS (bf16 pairs, fragment order) ----
    auto stage = [&](int k0, int buf) {
        for (int i = threadIdx.x; i < 512; i += 192) {
            int kp = i >> 5, col = i & 31;     // pair index, pixel in strip
            int colg = nb + col;
            int k = k0 + 2 * kp;
            float lo = Xb[(size_t)k * HW + colg];
            float hi = Xb[(size_t)(k + 1) * HW + colg];
            int t = col >> 4, lc = col & 15;
            int lane_s, d;
            if (kp < 8) { lane_s = lc;      d = kp;     }
            else        { lane_s = lc + 16; d = kp - 8; }
            lds[buf][t * 256 + d * 32 + lane_s] = pack_bf16(lo, hi);
        }
    };

    f32x8 acc[2] = {{}, {}};
    stage(0, 0);
    for (int c = 0; c < nchunks; ++c) {
        __syncthreads();
        // A fragment (weights) from global (tiny, L2-resident)
        bf16x16 a;
        const float* wr = W + (size_t)mrow * Cin + c * 32;
#pragma unroll
        for (int d = 0; d < 4; ++d) {
            int ka = koffA + 2 * d;
            float2 w0 = *(const float2*)(wr + ka);
            float2 w1 = *(const float2*)(wr + ka + 8);
            a[2 * d]         = (__bf16)w0.x;
            a[2 * d + 1]     = (__bf16)w0.y;
            a[8 + 2 * d]     = (__bf16)w1.x;
            a[8 + 2 * d + 1] = (__bf16)w1.y;
        }
        // B fragments from LDS (shared across the 6 waves)
        const int buf = c & 1;
#pragma unroll
        for (int t = 0; t < 2; ++t) {
            u32x8 raw;
#pragma unroll
            for (int d = 0; d < 8; ++d)
                raw[d] = lds[buf][t * 256 + d * 32 + lane];
            bf16x16 bm = __builtin_bit_cast(bf16x16, raw);
            acc[t] = __builtin_amdgcn_wmma_f32_16x16x32_bf16(
                false, a, false, bm, (short)0, acc[t], false, false);
        }
        if (c + 1 < nchunks) stage((c + 1) * 32, (c + 1) & 1);
    }

    const int m0 = mbase + ((lane < 16) ? 0 : 8);
#pragma unroll
    for (int t = 0; t < 2; ++t) {
        const int nc = nb + t * 16 + l15;
#pragma unroll
        for (int d = 0; d < 8; ++d) {
            int m   = m0 + d;
            float v = acc[t][d];
            if (BIASRELU) { v += bias[m]; v = fmaxf(v, 0.f); }
            size_t oi = (size_t)m * HW + nc;
            if (ADDSRC) v += addsrc[(size_t)b * Cout * HW + oi];
            Ob[oi] = v;
        }
    }
}

// =====================================================================
// Depthwise 3x3 (middle kd=1 slice of the 3x3x3 kernel on depth-1 volume)
// =====================================================================
__global__ __launch_bounds__(256)
void dw3x3(const float* __restrict__ in, const float* __restrict__ w,
           float* __restrict__ out)
{
    size_t i = (size_t)blockIdx.x * blockDim.x + threadIdx.x;
    size_t total = (size_t)BB * C3 * HW;
    if (i >= total) return;
    int p  = (int)(i % HW);
    int ch = (int)((i / HW) % C3);
    int y = p >> 7, x = p & 127;
    const float* wp = w + ch * 27 + 9;   // [ch,0,kd=1,ky,kx]
    const float* ip = in + (i - p);      // base of this (b,ch) plane
    float s = 0.f;
#pragma unroll
    for (int ky = 0; ky < 3; ++ky) {
        int yy = y + ky - 1;
        if ((unsigned)yy >= 128u) continue;
#pragma unroll
        for (int kx = 0; kx < 3; ++kx) {
            int xx = x + kx - 1;
            if ((unsigned)xx >= 128u) continue;
            s += ip[yy * 128 + xx] * wp[ky * 3 + kx];
        }
    }
    out[i] = s;
}

// =====================================================================
// fc-mix: the reshape(b,hw,12,48) scramble + [9x12] mix, producing
// f_conv[b, cc*9+j, p].  Flat view: Q[b][p*576 + g*48 + cc].
// =====================================================================
__global__ __launch_bounds__(256)
void fcmix(const float* __restrict__ Q, const float* __restrict__ fcw,
           const float* __restrict__ fcb, float* __restrict__ fconv)
{
    int n = blockIdx.x * blockDim.x + threadIdx.x;  // pixel
    int b = blockIdx.y;
    const float* Qb = Q + (size_t)b * C3 * HW;
    float*       Fb = fconv + (size_t)b * CMIX * HW;
    for (int cc = 0; cc < 48; ++cc) {
        float qg[12];
#pragma unroll
        for (int g = 0; g < 12; ++g)
            qg[g] = Qb[(size_t)n * 576 + g * 48 + cc];
#pragma unroll
        for (int j = 0; j < 9; ++j) {
            float s = fcb[j];
#pragma unroll
            for (int g = 0; g < 12; ++g) s += qg[g] * fcw[j * 12 + g];
            Fb[(size_t)(cc * 9 + j) * HW + n] = s;
        }
    }
}

// =====================================================================
// Grouped dep-conv: 48 groups, 9 in / 4 out per group, 3x3 middle slice
// =====================================================================
__global__ __launch_bounds__(256)
void depconv(const float* __restrict__ fconv, const float* __restrict__ w,
             const float* __restrict__ bias, float* __restrict__ out)
{
    size_t i = (size_t)blockIdx.x * blockDim.x + threadIdx.x;
    size_t total = (size_t)BB * CC0 * HW;
    if (i >= total) return;
    int p = (int)(i % HW);
    int o = (int)((i / HW) % CC0);
    int b = (int)(i / ((size_t)HW * CC0));
    int y = p >> 7, x = p & 127;
    int grp = o >> 2;
    const float* fb = fconv + ((size_t)b * CMIX + grp * 9) * HW;
    float s = bias[o];
    for (int j = 0; j < 9; ++j) {
        const float* ip = fb + (size_t)j * HW;
        const float* wp = w + o * 243 + j * 27 + 9;  // [o,j,kd=1,ky,kx]
#pragma unroll
        for (int ky = 0; ky < 3; ++ky) {
            int yy = y + ky - 1;
            if ((unsigned)yy >= 128u) continue;
#pragma unroll
            for (int kx = 0; kx < 3; ++kx) {
                int xx = x + kx - 1;
                if ((unsigned)xx >= 128u) continue;
                s += ip[yy * 128 + xx] * wp[ky * 3 + kx];
            }
        }
    }
    out[i] = s;
}

// =====================================================================
// Row L2-norm reciprocals for q (rows 0..767) and k (rows 768..1535)
// =====================================================================
__global__ __launch_bounds__(32)
void rownorm(const float* __restrict__ qkv, float* __restrict__ inv)
{
    int r    = blockIdx.x;
    int lane = threadIdx.x;
    int tensor = r / 768;
    int rem    = r % 768;
    int b = rem / 192, c = rem % 192;
    const float* row = qkv + ((size_t)b * C3 + tensor * 192 + c) * HW;
    float s = 0.f;
    for (int i = lane; i < HW; i += 32) { float v = row[i]; s += v * v; }
#pragma unroll
    for (int off = 16; off; off >>= 1) s += __shfl_xor(s, off, 32);
    if (lane == 0) {
        float nrm = sqrtf(s);
        inv[r] = 1.0f / fmaxf(nrm, 1e-12f);
    }
}

// =====================================================================
// attn += (q/|q|) . (k/|k|)^T   (K-split, atomic fp32 accumulation)
// grid: (b*h = 16, mt*3+nt = 9, ks = KSPLIT); one wave each.
// =====================================================================
__global__ __launch_bounds__(32)
void attn_qk(const float* __restrict__ qkv, const float* __restrict__ inv,
             float* __restrict__ attn)
{
    int bh = blockIdx.x;
    int b = bh >> 2, h = bh & 3;
    int mt = blockIdx.y / 3, nt = blockIdx.y % 3;
    int ks = blockIdx.z;
    int lane = threadIdx.x;
    int l15  = lane & 15;
    int koffA = (lane < 16) ? 0 : 16;
    int koffB = (lane < 16) ? 0 : 16;
    int qc = h * 48 + mt * 16 + l15;
    int kc = h * 48 + nt * 16 + l15;
    const float* qrow = qkv + ((size_t)b * C3 + qc) * HW;
    const float* krow = qkv + ((size_t)b * C3 + 192 + kc) * HW;
    float iq = inv[b * 192 + qc];
    float ik = inv[768 + b * 192 + kc];

    const int kbeg = ks * (HW / KSPLIT);
    const int kend = kbeg + (HW / KSPLIT);

    f32x8 acc = {};
    for (int k0 = kbeg; k0 < kend; k0 += 32) {
        bf16x16 a, bm;
#pragma unroll
        for (int d = 0; d < 4; ++d) {
            int ka = k0 + koffA + 2 * d;
            float2 q0 = *(const float2*)(qrow + ka);
            float2 q1 = *(const float2*)(qrow + ka + 8);
            a[2 * d]         = (__bf16)(q0.x * iq);
            a[2 * d + 1]     = (__bf16)(q0.y * iq);
            a[8 + 2 * d]     = (__bf16)(q1.x * iq);
            a[8 + 2 * d + 1] = (__bf16)(q1.y * iq);
        }
#pragma unroll
        for (int d = 0; d < 8; ++d) {
            float2 k2 = *(const float2*)(krow + k0 + koffB + 2 * d);
            bm[2 * d]     = (__bf16)(k2.x * ik);
            bm[2 * d + 1] = (__bf16)(k2.y * ik);
        }
        acc = __builtin_amdgcn_wmma_f32_16x16x32_bf16(false, a, false, bm,
                                                      (short)0, acc, false, false);
    }
    int   m0 = mt * 16 + ((lane < 16) ? 0 : 8);
    int   nn = nt * 16 + l15;
    float* ab = attn + (size_t)bh * 48 * 48;
#pragma unroll
    for (int d = 0; d < 8; ++d)
        atomicAdd(&ab[(size_t)(m0 + d) * 48 + nn], acc[d]);
}

// =====================================================================
// Gate: per-pixel sigmoid(w2 . relu_g1 + b2), global mean accumulation
// =====================================================================
__global__ __launch_bounds__(256)
void gate_reduce(const float* __restrict__ g1, const float* __restrict__ w2,
                 const float* __restrict__ b2, float* __restrict__ gsum)
{
    int p = blockIdx.x * 256 + threadIdx.x;
    int b = blockIdx.y;
    const float* gb = g1 + (size_t)b * 96 * HW;
    float s = b2[0];
    for (int o = 0; o < 96; ++o) s += gb[(size_t)o * HW + p] * w2[o];
    float g = 1.0f / (1.0f + __expf(-s));
    if (g != g) g = 0.f;  // nan_to_num
#pragma unroll
    for (int off = 16; off; off >>= 1) g += __shfl_xor(g, off, 32);
    __shared__ float wsum[8];
    int lane = threadIdx.x & 31, wv = threadIdx.x >> 5;
    if (lane == 0) wsum[wv] = g;
    __syncthreads();
    if (threadIdx.x == 0) {
        float t = 0.f;
        for (int i = 0; i < 8; ++i) t += wsum[i];
        atomicAdd(gsum, t);
    }
}

__global__ void zero_buf(float* __restrict__ p, int n)
{
    int i = blockIdx.x * blockDim.x + threadIdx.x;
    if (i < n) p[i] = 0.f;
}

__global__ void calc_k(const float* __restrict__ gsum, int* __restrict__ dynk)
{
    if (threadIdx.x == 0) {
        float mean = gsum[0] / (float)((size_t)BB * HW);
        int k = (int)floorf(48.0f * mean);
        if (k < 0) k = 0;
        if (k > 48) k = 48;
        dynk[0] = k;
    }
}

// =====================================================================
// temperature scale + top-k (argsort-stable rank) mask + softmax (48)
// =====================================================================
__global__ __launch_bounds__(64)
void topk_softmax(const float* __restrict__ attn, const int* __restrict__ dynk,
                  const float* __restrict__ temp, float* __restrict__ out)
{
    __shared__ float a[48], e[48];
    int r = blockIdx.x;
    int t = threadIdx.x;
    int h = (r / 48) & 3;
    const float* row = attn + (size_t)r * 48;
    if (t < 48) a[t] = row[t] * temp[h];
    __syncthreads();
    int K = dynk[0];
    float m = -__builtin_inff();
    if (t < 48) {
        float ai = a[t];
        int rank = 0;
        for (int j = 0; j < 48; ++j) {
            float aj = a[j];
            rank += (aj > ai) || (aj == ai && j < t);
        }
        m = (rank < K) ? ai : -__builtin_inff();
        e[t] = m;
    }
    __syncthreads();
    float mx = -__builtin_inff();
    if (t < 48) for (int j = 0; j < 48; ++j) mx = fmaxf(mx, e[j]);
    __syncthreads();
    if (t < 48) a[t] = (m == -__builtin_inff() || mx == -__builtin_inff())
                           ? 0.f : __expf(m - mx);
    __syncthreads();
    if (t < 48) {
        float sm = 0.f;
        for (int j = 0; j < 48; ++j) sm += a[j];
        out[(size_t)r * 48 + t] = (sm > 0.f) ? a[t] / sm : 0.f;
    }
}

// =====================================================================
// out = P(48x48) . v(48xHW) per (b,h) via WMMA; K=48 padded to 64
// =====================================================================
__global__ __launch_bounds__(256)
void attn_v(const float* __restrict__ P, const float* __restrict__ qkv,
            float* __restrict__ out)
{
    int bh = blockIdx.x;
    int b = bh >> 2, h = bh & 3;
    int mt = blockIdx.y;
    int wave = threadIdx.x >> 5, lane = threadIdx.x & 31, l15 = lane & 15;
    int nbase = (blockIdx.z * 8 + wave) * 16;
    int koffA = (lane < 16) ? 0 : 16;
    int koffB = (lane < 16) ? 0 : 16;
    const float* Pr = P + ((size_t)bh * 48 + mt * 16 + l15) * 48;
    const float* vb = qkv + ((size_t)b * C3 + 384 + h * 48) * HW;
    int nc = nbase + l15;

    f32x8 acc = {};
#pragma unroll
    for (int k0 = 0; k0 < 64; k0 += 32) {
        bf16x16 a, bm;
#pragma unroll
        for (int d = 0; d < 4; ++d) {
            int ka = k0 + koffA + 2 * d;
            a[2 * d]     = (__bf16)((ka     < 48) ? Pr[ka]     : 0.f);
            a[2 * d + 1] = (__bf16)((ka + 1 < 48) ? Pr[ka + 1] : 0.f);
            int ka2 = ka + 8;
            a[8 + 2 * d]     = (__bf16)((ka2     < 48) ? Pr[ka2]     : 0.f);
            a[8 + 2 * d + 1] = (__bf16)((ka2 + 1 < 48) ? Pr[ka2 + 1] : 0.f);
        }
#pragma unroll
        for (int d = 0; d < 8; ++d) {
            int kb = k0 + koffB + 2 * d;
            bm[2 * d]     = (__bf16)((kb     < 48) ? vb[(size_t)kb * HW + nc]       : 0.f);
            bm[2 * d + 1] = (__bf16)((kb + 1 < 48) ? vb[(size_t)(kb + 1) * HW + nc] : 0.f);
        }
        acc = __builtin_amdgcn_wmma_f32_16x16x32_bf16(false, a, false, bm,
                                                      (short)0, acc, false, false);
    }
    int m0 = mt * 16 + ((lane < 16) ? 0 : 8);
    float* ob = out + ((size_t)b * CC0 + h * 48) * HW;
#pragma unroll
    for (int d = 0; d < 8; ++d) ob[(size_t)(m0 + d) * HW + nc] = acc[d];
}

// =====================================================================
extern "C" void kernel_launch(void* const* d_in, const int* in_sizes, int n_in,
                              void* d_out, int out_size, void* d_ws, size_t ws_size,
                              hipStream_t stream)
{
    const float* x       = (const float*)d_in[0];
    const float* temp    = (const float*)d_in[1];
    const float* qkv_w   = (const float*)d_in[2];   // [576,192]
    const float* qkv_dww = (const float*)d_in[3];   // [576,1,3,3,3]
    const float* proj_w  = (const float*)d_in[4];   // [192,192]
    const float* fc_w    = (const float*)d_in[5];   // [9,12]
    const float* fc_b    = (const float*)d_in[6];   // [9]
    const float* dep_w   = (const float*)d_in[7];   // [192,9,3,3,3]
    const float* dep_b   = (const float*)d_in[8];   // [192]
    const float* gate_w1 = (const float*)d_in[9];   // [96,192]
    const float* gate_b1 = (const float*)d_in[10];  // [96]
    const float* gate_w2 = (const float*)d_in[11];  // [96]
    const float* gate_b2 = (const float*)d_in[12];  // [1]
    float* out = (float*)d_out;

    // ---- workspace layout (floats) ----
    float* ws      = (float*)d_ws;
    float* qkvraw  = ws;                     // [4,576,HW] 37,748,736 f (region R0)
    float* fconv   = ws;                     // reuse R0 after dw3x3 (28,311,552 f)
    float* attnout = ws;                     // reuse R0 after depconv (12,582,912 f)
    float* g1      = ws + 30000000;          // [4,96,HW] 6,291,456 f (R0 tail)
    float* qkvdw   = ws + 37748736;          // [4,576,HW] 37,748,736 f
    float* smallb  = ws + 75497472;
    float* invn    = smallb;                 // 1536 (q then k)
    float* attn    = smallb + 2048;          // 16*48*48 = 36864
    float* attnP   = attn + 36864;           // 36864
    float* gsum    = attnP + 36864;          // 1
    int*   dynk    = (int*)(gsum + 1);       // 1

    // 0) zero accumulators (ws is not re-zeroed between graph replays)
    zero_buf<<<1, 32, 0, stream>>>(gsum, 1);
    zero_buf<<<(36864 + 255) / 256, 256, 0, stream>>>(attn, 36864);

    // 1) qkv pointwise GEMM: [576 x 192] x [192 x HW] per batch
    pw_gemm_wmma<0, 0><<<dim3(BB, C3 / 96, HW / 32), 192, 0, stream>>>(
        x, qkv_w, nullptr, nullptr, qkvraw, CC0, C3);

    // 2) depthwise 3x3 (middle slice of 3x3x3)
    {
        size_t total = (size_t)BB * C3 * HW;
        dw3x3<<<(unsigned)((total + 255) / 256), 256, 0, stream>>>(qkvraw, qkv_dww, qkvdw);
    }

    // 3) fc mix -> f_conv  (qkvraw dead; R0 reused)
    fcmix<<<dim3(HW / 256, BB), 256, 0, stream>>>(qkvdw, fc_w, fc_b, fconv);

    // 4) grouped dep-conv -> out_conv written directly to d_out
    {
        size_t total = (size_t)BB * CC0 * HW;
        depconv<<<(unsigned)((total + 255) / 256), 256, 0, stream>>>(fconv, dep_w, dep_b, out);
    }

    // 5) gate stage 1 GEMM (WMMA, fused bias + relu)
    pw_gemm_wmma<1, 0><<<dim3(BB, 96 / 96, HW / 32), 192, 0, stream>>>(
        x, gate_w1, gate_b1, nullptr, g1, CC0, 96);

    // 6) gate stage 2 + global mean accumulation
    gate_reduce<<<dim3(HW / 256, BB), 256, 0, stream>>>(g1, gate_w2, gate_b2, gsum);

    // 7) dynamic_k
    calc_k<<<1, 1, 0, stream>>>(gsum, dynk);

    // 8) L2 norms of q and k rows
    rownorm<<<1536, 32, 0, stream>>>(qkvdw, invn);

    // 9) attn += q . k^T   (WMMA, K-split 16, atomic accumulate)
    attn_qk<<<dim3(BB * HEADS, 9, KSPLIT), 32, 0, stream>>>(qkvdw, invn, attn);

    // 10) temperature + top-k mask + softmax
    topk_softmax<<<768, 64, 0, stream>>>(attn, dynk, temp, attnP);

    // 11) out = attn . v  (WMMA, K = 48 padded to 64) -> attnout (reuses R0)
    attn_v<<<dim3(BB * HEADS, 3, HW / 128), 256, 0, stream>>>(attnP, qkvdw, attnout);

    // 12) proj GEMM + add out_conv (already in d_out) -> d_out
    pw_gemm_wmma<0, 1><<<dim3(BB, CC0 / 96, HW / 32), 192, 0, stream>>>(
        attnout, proj_w, nullptr, out, out, CC0, CC0);
}
